// LocMotionAppearance_86801289052825
// MI455X (gfx1250) — compile-verified, matrix operand order checked by reading, not run on hardware
//
#include <hip/hip_runtime.h>

typedef __attribute__((ext_vector_type(16))) __bf16 v16bf;
typedef __attribute__((ext_vector_type(8)))  __bf16 v8bf;
typedef __attribute__((ext_vector_type(8)))  float  v8f;

#define DEVFN static __device__ __forceinline__

constexpr int  Bn   = 8;
constexpr int  NSP  = 1024;
constexpr int  Hh   = 256;
constexpr int  Ww   = 256;
constexpr int  Nn   = Bn * NSP;        // 8192 superpixel nodes
constexpr int  En   = 131072;          // edges
constexpr float MIXC = 0.5f;
constexpr float INV_SIGMA = 20.0f;     // 1/0.05

DEVFN void atomAddF(float* p, float v) { unsafeAtomicAdd(p, v); }

// ---------------------------------------------------------------------------
// Per-pixel segment statistics: counts and coordinate sums (for pc)
// ---------------------------------------------------------------------------
__global__ __launch_bounds__(256) void seg_stats_kernel(
    const int* __restrict__ labels, float* __restrict__ cnt,
    float* __restrict__ pcsum) {
  int p = blockIdx.x * 256 + threadIdx.x;          // 0 .. B*H*W-1
  int b  = p >> 16;
  int yx = p & 65535;
  int y  = yx >> 8;
  int x  = yx & 255;
  int n  = labels[p] + (b << 10);
  atomAddF(&cnt[n], 1.0f);
  atomAddF(&pcsum[2 * n + 0], (float)y * (1.0f / 255.0f));   // channel 0: row
  atomAddF(&pcsum[2 * n + 1], (float)x * (1.0f / 255.0f));   // channel 1: col
}

__global__ __launch_bounds__(256) void finalize_pc_kernel(
    const float* __restrict__ cnt, const float* __restrict__ pcsum,
    float* __restrict__ pc) {
  int n = blockIdx.x * 256 + threadIdx.x;
  if (n >= Nn) return;
  float c = cnt[n] > 1.0f ? cnt[n] : 1.0f;
  pc[2 * n + 0] = pcsum[2 * n + 0] / c;
  pc[2 * n + 1] = pcsum[2 * n + 1] / c;
}

__global__ __launch_bounds__(256) void edge_attr_kernel(
    const int* __restrict__ src, const int* __restrict__ dst,
    const float* __restrict__ pc, float* __restrict__ ea) {
  int e = blockIdx.x * 256 + threadIdx.x;
  if (e >= En) return;
  int s = src[e], d = dst[e];
  float dx = pc[2 * s + 0] - pc[2 * d + 0];
  float dy = pc[2 * s + 1] - pc[2 * d + 1];
  ea[e] = __expf(-(dx * dx + dy * dy) * INV_SIGMA);
}

// ---------------------------------------------------------------------------
// Bilinear pooling weight matrix (one frame, one skip level):
//   Wm[superpixel, input_cell] += bilinear weight of cell at each pixel
// ---------------------------------------------------------------------------
__global__ __launch_bounds__(256) void build_wmat_kernel(
    const int* __restrict__ labels_frame, float* __restrict__ Wm,
    int hin, int win) {
  int p = blockIdx.x * 256 + threadIdx.x;          // 0..65535 within frame
  int y = p >> 8;
  int x = p & 255;
  int s = labels_frame[p];                          // 0..1023
  float py = (float)y * ((float)(hin - 1) / 255.0f);
  float px = (float)x * ((float)(win - 1) / 255.0f);
  int iy0 = (int)py;  if (iy0 > hin - 1) iy0 = hin - 1;
  int ix0 = (int)px;  if (ix0 > win - 1) ix0 = win - 1;
  int iy1 = iy0 + 1 < hin ? iy0 + 1 : hin - 1;
  int ix1 = ix0 + 1 < win ? ix0 + 1 : win - 1;
  float wy = py - (float)iy0;
  float wx = px - (float)ix0;
  float* row = Wm + (size_t)s * (size_t)(hin * win);
  atomAddF(&row[iy0 * win + ix0], (1.0f - wy) * (1.0f - wx));
  atomAddF(&row[iy0 * win + ix1], (1.0f - wy) * wx);
  atomAddF(&row[iy1 * win + ix0], wy * (1.0f - wx));
  atomAddF(&row[iy1 * win + ix1], wy * wx);
}

// ---------------------------------------------------------------------------
// NT GEMM with bf16 WMMA, f32 accumulate:  C[M,N] = A[M,K] * Bt[N,K]^T
// A, Bt are f32 row-major with K contiguous; converted to bf16 in LDS staging.
// Requires M%64==0, N%64==0, K%32==0 (all call sites satisfy this).
// Block = 128 threads (4 waves); 64x64 tile; wave w owns rows [16w,16w+16).
// All 5 fragment loads are hoisted ahead of the 4 WMMAs so the ds_loads
// clause under one s_wait_dscnt and the WMMAs issue back-to-back.
// ---------------------------------------------------------------------------
__global__ __launch_bounds__(128) void gemm_nt_bf16_wmma(
    const float* __restrict__ A, const float* __restrict__ Bt,
    float* __restrict__ C, int M, int N, int K) {
  __shared__ __align__(16) __bf16 lA[64 * 32];
  __shared__ __align__(16) __bf16 lB[64 * 32];
  const int mblk = blockIdx.x * 64;
  const int nblk = blockIdx.y * 64;
  const int tid  = threadIdx.x;
  const int wave = tid >> 5;
  const int lane = tid & 31;
  const int r    = lane & 15;       // row within 16-row fragment / N within C tile
  const int half = lane >> 4;       // K-half selector per ISA A-layout

  v8f acc[4];
  acc[0] = v8f{}; acc[1] = v8f{}; acc[2] = v8f{}; acc[3] = v8f{};

  const int lrow = tid >> 1;        // 0..63 : tile row this thread stages
  const int lcol = (tid & 1) * 16;  // 0 or 16 : K offset this thread stages
  const float* aptr = A  + (size_t)(mblk + lrow) * (size_t)K + lcol;
  const float* bptr = Bt + (size_t)(nblk + lrow) * (size_t)K + lcol;

  union Frag { v16bf v; v8bf h[2]; };

  for (int kc = 0; kc < K; kc += 32) {
    // prefetch next K-chunk global strips into cache (global_prefetch_b8)
    if (kc + 32 < K) {
      __builtin_prefetch((const void*)(aptr + kc + 32), 0, 1);
      __builtin_prefetch((const void*)(bptr + kc + 32), 0, 1);
    }
    __syncthreads();
    const float4* a4 = (const float4*)(aptr + kc);
    const float4* b4 = (const float4*)(bptr + kc);
#pragma unroll
    for (int i = 0; i < 4; ++i) {
      float4 av = a4[i];
      float4 bv = b4[i];
      int o = lrow * 32 + lcol + i * 4;
      lA[o + 0] = (__bf16)av.x; lA[o + 1] = (__bf16)av.y;
      lA[o + 2] = (__bf16)av.z; lA[o + 3] = (__bf16)av.w;
      lB[o + 0] = (__bf16)bv.x; lB[o + 1] = (__bf16)bv.y;
      lB[o + 2] = (__bf16)bv.z; lB[o + 3] = (__bf16)bv.w;
    }
    __syncthreads();

    // A fragment: lane L holds row (L&15); K-halves per ISA 16-bit A layout:
    // half=0 -> K{0..7,16..23}, half=1 -> K{8..15,24..31}
    const __bf16* arow = &lA[(wave * 16 + r) * 32];
    Frag af;
    af.h[0] = *(const v8bf*)(arow + half * 8);
    af.h[1] = *(const v8bf*)(arow + 16 + half * 8);

    Frag bfrag[4];
#pragma unroll
    for (int j = 0; j < 4; ++j) {
      const __bf16* brow = &lB[(j * 16 + r) * 32];
      bfrag[j].h[0] = *(const v8bf*)(brow + half * 8);
      bfrag[j].h[1] = *(const v8bf*)(brow + 16 + half * 8);
    }
#pragma unroll
    for (int j = 0; j < 4; ++j) {
      acc[j] = __builtin_amdgcn_wmma_f32_16x16x32_bf16(
          false, af.v, false, bfrag[j].v, (short)0, acc[j], false, false);
    }
  }

  // C/D layout: lane 0-15 -> N=lane, VGPR v -> M=v ; lanes 16-31 -> M=8+v
#pragma unroll
  for (int j = 0; j < 4; ++j) {
    int n = nblk + j * 16 + r;
#pragma unroll
    for (int v = 0; v < 8; ++v) {
      int m = mblk + wave * 16 + half * 8 + v;
      C[(size_t)m * (size_t)N + n] = acc[j][v];
    }
  }
}

// pooled[n][c] /= max(cnt[n],1)
__global__ __launch_bounds__(256) void norm_pooled_kernel(
    float* __restrict__ P, const float* __restrict__ cnt, int C) {
  int idx = blockIdx.x * 256 + threadIdx.x;
  int n = idx / C;
  float c = cnt[n] > 1.0f ? cnt[n] : 1.0f;
  P[idx] /= c;
}

// Wt[n*K + k] = W[k*256 + n]   (W is [K,256] row-major)
__global__ __launch_bounds__(256) void transpose_w_kernel(
    const float* __restrict__ W, float* __restrict__ Wt, int K) {
  int idx = blockIdx.x * 256 + threadIdx.x;   // K*256 threads
  int k = idx >> 8;
  int n = idx & 255;
  Wt[(size_t)n * K + k] = W[idx];
}

// agg[dst][c] += ew[e] * h[src][c] ; one thread = one edge x 4 channels
__global__ __launch_bounds__(256) void aggregate_kernel(
    const float* __restrict__ h, const int* __restrict__ src,
    const int* __restrict__ dst, const float* __restrict__ ea,
    float* __restrict__ agg) {
  int gid = blockIdx.x * 256 + threadIdx.x;
  int e  = gid >> 6;
  int c0 = (gid & 63) * 4;
  float w = ea[e];
  int s = src[e], d = dst[e];
  const float4 hv = *(const float4*)(h + (size_t)s * 256 + c0);
  float* ap = agg + (size_t)d * 256 + c0;
  atomAddF(ap + 0, w * hv.x);
  atomAddF(ap + 1, w * hv.y);
  atomAddF(ap + 2, w * hv.z);
  atomAddF(ap + 3, w * hv.w);
}

__global__ __launch_bounds__(256) void bias_relu_kernel(
    const float* __restrict__ agg, const float* __restrict__ b,
    float* __restrict__ x) {
  int idx = blockIdx.x * 256 + threadIdx.x;   // N*256 threads
  float v = agg[idx] + b[idx & 255];
  x[idx] = v > 0.0f ? v : 0.0f;
}

__global__ __launch_bounds__(256) void mix_kernel(
    const float* __restrict__ x, const float* __restrict__ pooled,
    float* __restrict__ xm) {
  int idx = blockIdx.x * 256 + threadIdx.x;   // N*256 threads
  xm[idx] = (1.0f - MIXC) * x[idx] + MIXC * pooled[idx];
}

// s[n] = sum_c x[n][c] * lin_w[c]  ; one wave per node (wave32)
__global__ __launch_bounds__(256) void score_kernel(
    const float* __restrict__ x, const float* __restrict__ w,
    float* __restrict__ s) {
  int wave = threadIdx.x >> 5;
  int lane = threadIdx.x & 31;
  int n = blockIdx.x * 8 + wave;
  const float* xr = x + (size_t)n * 256;
  float acc = 0.0f;
#pragma unroll
  for (int c = lane; c < 256; c += 32) acc += xr[c] * w[c];
#pragma unroll
  for (int off = 16; off > 0; off >>= 1) acc += __shfl_down(acc, off, 32);
  if (lane == 0) s[n] = acc;
}

DEVFN float sigmoidf(float x) { return 1.0f / (1.0f + __expf(-x)); }

__global__ __launch_bounds__(256) void pairs_kernel(
    const float* __restrict__ s, const int* __restrict__ src,
    const int* __restrict__ dst, const int* __restrict__ negs,
    float* __restrict__ out) {
  int e = blockIdx.x * 256 + threadIdx.x;
  if (e >= En) return;
  float ss = s[src[e]];
  out[e]      = sigmoidf(ss - s[negs[e]]);   // dan
  out[En + e] = sigmoidf(ss - s[dst[e]]);    // dap
}

// ---------------------------------------------------------------------------
extern "C" void kernel_launch(void* const* d_in, const int* in_sizes, int n_in,
                              void* d_out, int out_size, void* d_ws, size_t ws_size,
                              hipStream_t stream) {
  (void)in_sizes; (void)n_in; (void)out_size; (void)ws_size;

  const int*   labels = (const int*)  d_in[0];
  const float* skip0  = (const float*)d_in[1];   // [8,512,32,32]
  const float* skip1  = (const float*)d_in[2];   // [8,256,64,64]
  const float* skip2  = (const float*)d_in[3];   // [8,256,128,128]
  const int*   edges  = (const int*)  d_in[4];   // [2,E]
  const int*   negs   = (const int*)  d_in[5];
  const float* W0     = (const float*)d_in[6];   // [512,256]
  const float* b0     = (const float*)d_in[7];
  const float* W1     = (const float*)d_in[8];   // [256,256]
  const float* b1     = (const float*)d_in[9];
  const float* W2     = (const float*)d_in[10];  // [256,256]
  const float* b2     = (const float*)d_in[11];
  const float* lin_w  = (const float*)d_in[12];  // [256]
  float* out = (float*)d_out;

  const int* src = edges;
  const int* dst = edges + En;

  // ---- workspace carve-out (f32 elements) ----
  float* ws = (float*)d_ws;
  size_t off = 0;
  auto alloc = [&](size_t n) { float* p = ws + off; off += n; return p; };
  float* cnt     = alloc(Nn);
  float* pcsum   = alloc(2 * Nn);         // contiguous with cnt (joint memset)
  float* pc      = alloc(2 * Nn);
  float* pooled0 = alloc((size_t)Nn * 512);
  float* pooled1 = alloc((size_t)Nn * 256);
  float* pooled2 = alloc((size_t)Nn * 256);
  float* Wmat    = alloc((size_t)1024 * 16384);  // reused per (frame, level)
  float* W0t     = alloc((size_t)256 * 512);
  float* W1t     = alloc((size_t)256 * 256);
  float* W2t     = alloc((size_t)256 * 256);
  float* hbuf    = alloc((size_t)Nn * 256);
  float* aggb    = alloc((size_t)Nn * 256);
  float* xbuf    = alloc((size_t)Nn * 256);
  float* xmix    = alloc((size_t)Nn * 256);
  float* sbuf    = alloc(Nn);

  // ---- segment stats + pooled coords + edge weights ----
  hipMemsetAsync(cnt, 0, (size_t)3 * Nn * sizeof(float), stream);  // cnt+pcsum
  seg_stats_kernel<<<(Bn * Hh * Ww) / 256, 256, 0, stream>>>(labels, cnt, pcsum);
  finalize_pc_kernel<<<Nn / 256, 256, 0, stream>>>(cnt, pcsum, pc);
  edge_attr_kernel<<<En / 256, 256, 0, stream>>>(src, dst, pc, out + 2 * (size_t)En);
  float* ea = out + 2 * (size_t)En;

  // ---- superpixel pooling as (bilinear weight matrix) x (features^T) GEMM ----
  const int   hins[3]  = {32, 64, 128};
  const int   chans[3] = {512, 256, 256};
  const float* skips[3] = {skip0, skip1, skip2};
  float* pooleds[3] = {pooled0, pooled1, pooled2};
  for (int f = 0; f < Bn; ++f) {
    for (int l = 0; l < 3; ++l) {
      int hin = hins[l], cells = hin * hin, C = chans[l];
      hipMemsetAsync(Wmat, 0, (size_t)1024 * cells * sizeof(float), stream);
      build_wmat_kernel<<<(Hh * Ww) / 256, 256, 0, stream>>>(
          labels + (size_t)f * Hh * Ww, Wmat, hin, hin);
      dim3 grid(1024 / 64, C / 64);
      gemm_nt_bf16_wmma<<<grid, 128, 0, stream>>>(
          Wmat, skips[l] + (size_t)f * C * cells,
          pooleds[l] + (size_t)f * 1024 * C, 1024, C, cells);
    }
  }
  norm_pooled_kernel<<<(size_t)Nn * 512 / 256, 256, 0, stream>>>(pooled0, cnt, 512);
  norm_pooled_kernel<<<(size_t)Nn * 256 / 256, 256, 0, stream>>>(pooled1, cnt, 256);
  norm_pooled_kernel<<<(size_t)Nn * 256 / 256, 256, 0, stream>>>(pooled2, cnt, 256);

  // ---- weight transposes for NT GEMM ----
  transpose_w_kernel<<<(512 * 256) / 256, 256, 0, stream>>>(W0, W0t, 512);
  transpose_w_kernel<<<(256 * 256) / 256, 256, 0, stream>>>(W1, W1t, 256);
  transpose_w_kernel<<<(256 * 256) / 256, 256, 0, stream>>>(W2, W2t, 256);

  const size_t nbytes256 = (size_t)Nn * 256 * sizeof(float);
  const int nblk256 = (Nn * 256) / 256;

  // ---- GCN layer 0: x = relu(agg(pooled0 @ W0) + b0) ----
  gemm_nt_bf16_wmma<<<dim3(Nn / 64, 256 / 64), 128, 0, stream>>>(
      pooled0, W0t, hbuf, Nn, 256, 512);
  hipMemsetAsync(aggb, 0, nbytes256, stream);
  aggregate_kernel<<<(En * 64) / 256, 256, 0, stream>>>(hbuf, src, dst, ea, aggb);
  bias_relu_kernel<<<nblk256, 256, 0, stream>>>(aggb, b0, xbuf);

  // ---- GCN layer 1 ----
  mix_kernel<<<nblk256, 256, 0, stream>>>(xbuf, pooled1, xmix);
  gemm_nt_bf16_wmma<<<dim3(Nn / 64, 256 / 64), 128, 0, stream>>>(
      xmix, W1t, hbuf, Nn, 256, 256);
  hipMemsetAsync(aggb, 0, nbytes256, stream);
  aggregate_kernel<<<(En * 64) / 256, 256, 0, stream>>>(hbuf, src, dst, ea, aggb);
  bias_relu_kernel<<<nblk256, 256, 0, stream>>>(aggb, b1, xbuf);

  // ---- GCN layer 2 ----
  mix_kernel<<<nblk256, 256, 0, stream>>>(xbuf, pooled2, xmix);
  gemm_nt_bf16_wmma<<<dim3(Nn / 64, 256 / 64), 128, 0, stream>>>(
      xmix, W2t, hbuf, Nn, 256, 256);
  hipMemsetAsync(aggb, 0, nbytes256, stream);
  aggregate_kernel<<<(En * 64) / 256, 256, 0, stream>>>(hbuf, src, dst, ea, aggb);
  bias_relu_kernel<<<nblk256, 256, 0, stream>>>(aggb, b2, xbuf);

  // ---- scores + output sigmoids ----
  score_kernel<<<Nn / 8, 256, 0, stream>>>(xbuf, lin_w, sbuf);
  pairs_kernel<<<En / 256, 256, 0, stream>>>(sbuf, src, dst, negs, out);
}
